// MatchModel_13340168422013
// MI455X (gfx1250) — matrix operand order repaired and checked by reading, not compiled, not run on hardware
//
#include <hip/hip_runtime.h>
#include <hip/hip_bf16.h>
#include <cstdint>
#include <cstddef>

// ---- problem constants (match reference) ----
#define NUV   50000
#define NUSV  20000
#define NIV   100000
#define NISV  20000
#define EUS   200000
#define EIS   400000
#define EUI   500000
#define DE    64
#define NH    4
#define FHV   32
#define BB    1024

typedef __attribute__((ext_vector_type(16))) _Float16 v16h;
typedef __attribute__((ext_vector_type(8)))  float    v8f;

static inline int cdiv_h(long long a, long long b) { return (int)((a + b - 1) / b); }
static inline int gsz(long long n) {
  long long b = (n + 255) / 256;
  if (b > 1048576) b = 1048576;
  if (b < 1) b = 1;
  return (int)b;
}

// =============================== kernels ===============================

// vectorized gather: out[i,:] = table[ids[i],:], d multiple of 4
__global__ void k_gather4(const float* __restrict__ table, const int* __restrict__ ids,
                          float* __restrict__ out, int n, int d4) {
  long long tot = (long long)n * d4;
  for (long long i = (long long)blockIdx.x * blockDim.x + threadIdx.x; i < tot;
       i += (long long)gridDim.x * blockDim.x) {
    int row = (int)(i / d4), c = (int)(i % d4);
    ((float4*)out)[i] = ((const float4*)table)[(long long)ids[row] * d4 + c];
  }
}

__global__ void k_fill(float* __restrict__ p, float v, long long n) {
  for (long long i = (long long)blockIdx.x * blockDim.x + threadIdx.x; i < n;
       i += (long long)gridDim.x * blockDim.x)
    p[i] = v;
}

// out[i] = bias[i % cols], vectorized by 4 (cols multiple of 4)
__global__ void k_bias_bcast4(float* __restrict__ out, const float* __restrict__ bias,
                              long long n4, int cols4) {
  for (long long i = (long long)blockIdx.x * blockDim.x + threadIdx.x; i < n4;
       i += (long long)gridDim.x * blockDim.x)
    ((float4*)out)[i] = ((const float4*)bias)[(int)(i % cols4)];
}

// concat rows: out[n, da+db], da/db multiples of 4
__global__ void k_concat4(const float* __restrict__ a, const float* __restrict__ b,
                          float* __restrict__ out, int n, int da4, int db4) {
  int dt4 = da4 + db4;
  long long tot = (long long)n * dt4;
  for (long long i = (long long)blockIdx.x * blockDim.x + threadIdx.x; i < tot;
       i += (long long)gridDim.x * blockDim.x) {
    int row = (int)(i / dt4), c = (int)(i % dt4);
    ((float4*)out)[i] = (c < da4) ? ((const float4*)a)[(long long)row * da4 + c]
                                  : ((const float4*)b)[(long long)row * db4 + (c - da4)];
  }
}

// out[n,H] = einsum("nhf,hf->nh", f[n,H,fo], a[H,fo]); fo multiple of 4
__global__ void k_attn_dot(const float* __restrict__ f, const float* __restrict__ a,
                           float* __restrict__ out, int n, int fo) {
  long long tot = (long long)n * NH;
  int fo4 = fo >> 2;
  for (long long i = (long long)blockIdx.x * blockDim.x + threadIdx.x; i < tot;
       i += (long long)gridDim.x * blockDim.x) {
    int h = (int)(i % NH);
    long long row = i / NH;
    const float4* fp = (const float4*)(f + (row * NH + h) * (long long)fo);
    const float4* ap = (const float4*)(a + (long long)h * fo);
    float s = 0.f;
    for (int j = 0; j < fo4; ++j) {
      float4 x = fp[j], y = ap[j];
      s += x.x * y.x + x.y * y.y + x.z * y.z + x.w * y.w;
    }
    out[i] = s;
  }
}

// ee[e,h] = leaky_relu(el[es[e],h] + er[ed[e],h], 0.2); segment max into m[ed,h]
__global__ void k_edge_logits(const float* __restrict__ el, const float* __restrict__ er,
                              const int* __restrict__ es, const int* __restrict__ ed,
                              float* __restrict__ ee, float* __restrict__ m, int E) {
  long long tot = (long long)E * NH;
  for (long long i = (long long)blockIdx.x * blockDim.x + threadIdx.x; i < tot;
       i += (long long)gridDim.x * blockDim.x) {
    int h = (int)(i % NH);
    int e = (int)(i / NH);
    float v = el[(long long)es[e] * NH + h] + er[(long long)ed[e] * NH + h];
    v = v > 0.f ? v : 0.2f * v;
    ee[i] = v;
    float* addr = &m[(long long)ed[e] * NH + h];
    if (v >= 0.f) atomicMax((int*)addr, __float_as_int(v));
    else          atomicMin((unsigned int*)addr, __float_as_uint(v));
  }
}

__global__ void k_fix_m(float* __restrict__ m, long long n) {
  for (long long i = (long long)blockIdx.x * blockDim.x + threadIdx.x; i < n;
       i += (long long)gridDim.x * blockDim.x)
    if (!(m[i] > -1e37f)) m[i] = 0.f;   // empty segments (-inf) -> 0
}

// ex[e,h] = exp(ee - m[ed]); den[ed,h] += ex
__global__ void k_edge_exp(const float* __restrict__ ee, const float* __restrict__ m,
                           const int* __restrict__ ed, float* __restrict__ ex,
                           float* __restrict__ den, int E) {
  long long tot = (long long)E * NH;
  for (long long i = (long long)blockIdx.x * blockDim.x + threadIdx.x; i < tot;
       i += (long long)gridDim.x * blockDim.x) {
    int h = (int)(i % NH);
    int e = (int)(i / NH);
    float v = __expf(ee[i] - m[(long long)ed[e] * NH + h]);
    ex[i] = v;
    atomicAdd(&den[(long long)ed[e] * NH + h], v);
  }
}

// out[ed,h,f] += (ex/den) * fs[es,h,f]; 4 features per thread (fo multiple of 4)
__global__ void k_edge_aggr4(const float* __restrict__ ex, const float* __restrict__ den,
                             const float* __restrict__ fs,
                             const int* __restrict__ es, const int* __restrict__ ed,
                             float* __restrict__ out, int E, int fo) {
  int fo4 = fo >> 2;
  long long tot = (long long)E * NH * fo4;
  for (long long i = (long long)blockIdx.x * blockDim.x + threadIdx.x; i < tot;
       i += (long long)gridDim.x * blockDim.x) {
    int f4 = (int)(i % fo4);
    long long t = i / fo4;
    int h = (int)(t % NH);
    int e = (int)(t / NH);
    int dn = ed[e], sn = es[e];
    float a = ex[(long long)e * NH + h] / fmaxf(den[(long long)dn * NH + h], 1e-12f);
    float4 v = *(const float4*)&fs[((long long)sn * NH + h) * fo + 4 * f4];
    float* o = &out[((long long)dn * NH + h) * fo + 4 * f4];
    atomicAdd(o + 0, a * v.x);
    atomicAdd(o + 1, a * v.y);
    atomicAdd(o + 2, a * v.z);
    atomicAdd(o + 3, a * v.w);
  }
}

// ---------- WMMA GEMM with pre-packed f16 operands ----------
// Packed A layout: [M/16][K/32][lane 0..31][16 halves], half h of lane:
//   row = tm*16 + (lane&15); k = k0 + h + (h<8?0:8) + (lane>>4)*8   (ISA 16-bit A 16x32)
__global__ void k_pack_a(const float* __restrict__ A, _Float16* __restrict__ P,
                         int M, int K) {
  int ksteps = K >> 5;
  long long tot = (long long)(M >> 4) * ksteps * 32;
  for (long long i = (long long)blockIdx.x * blockDim.x + threadIdx.x; i < tot;
       i += (long long)gridDim.x * blockDim.x) {
    int lane = (int)(i & 31);
    long long t = i >> 5;
    int ks = (int)(t % ksteps);
    int tm = (int)(t / ksteps);
    int laneHi = lane >> 4, laneLo = lane & 15;
    const float4* src = (const float4*)(A + (long long)(tm * 16 + laneLo) * K + ks * 32 + laneHi * 8);
    float4 x0 = src[0], x1 = src[1];   // K = base .. base+7
    float4 y0 = src[4], y1 = src[5];   // K = base+16 .. base+23
    v16h o;
    o[0] = (_Float16)x0.x;  o[1] = (_Float16)x0.y;  o[2] = (_Float16)x0.z;  o[3] = (_Float16)x0.w;
    o[4] = (_Float16)x1.x;  o[5] = (_Float16)x1.y;  o[6] = (_Float16)x1.z;  o[7] = (_Float16)x1.w;
    o[8] = (_Float16)y0.x;  o[9] = (_Float16)y0.y;  o[10] = (_Float16)y0.z; o[11] = (_Float16)y0.w;
    o[12] = (_Float16)y1.x; o[13] = (_Float16)y1.y; o[14] = (_Float16)y1.z; o[15] = (_Float16)y1.w;
    ((v16h*)P)[i] = o;
  }
}

// Packed B layout: [N/16][K/32][lane][16 halves]; half h of lane:
//   col = tn*16 + (lane&15); k = k0 + h + (h<8?0:8) + (lane>>4)*8    (ISA 16-bit B 32x16)
__global__ void k_pack_b(const float* __restrict__ W, _Float16* __restrict__ P,
                         int N, int K) {
  int ksteps = K >> 5;
  long long tot = (long long)(N >> 4) * ksteps * 32;
  for (long long i = (long long)blockIdx.x * blockDim.x + threadIdx.x; i < tot;
       i += (long long)gridDim.x * blockDim.x) {
    int lane = (int)(i & 31);
    long long t = i >> 5;
    int ks = (int)(t % ksteps);
    int tn = (int)(t / ksteps);
    int laneHi = lane >> 4, laneLo = lane & 15;
    const float* src = W + (long long)(ks * 32 + laneHi * 8) * N + tn * 16 + laneLo;
    v16h o;
#pragma unroll
    for (int h = 0; h < 16; ++h) o[h] = (_Float16)src[(long long)(h + (h < 8 ? 0 : 8)) * N];
    ((v16h*)P)[i] = o;
  }
}

// C[M,N] = A@W (+bias,+relu). One wave computes a 16x64 strip: 4 n-tiles share one A load.
// Requires M%16==0, N%64==0, K%32==0 (true for all call sites: N in {64,128,256}).
__global__ void k_gemm_wmma(const _Float16* __restrict__ PA, const _Float16* __restrict__ PB,
                            const float* __restrict__ bias, float* __restrict__ C,
                            int M, int N, int K, int relu) {
  int wave = (int)((blockIdx.x * (long long)blockDim.x + threadIdx.x) >> 5);
  int lane = threadIdx.x & 31;
  int nQuads = N >> 6;
  int tiles  = (M >> 4) * nQuads;
  if (wave >= tiles) return;
  int tm = wave / nQuads;
  int tq = wave % nQuads;
  int ksteps = K >> 5;

  const v16h* pa  = (const v16h*)PA + (long long)tm * ksteps * 32 + lane;
  const v16h* pb0 = (const v16h*)PB + (long long)(tq * 4 + 0) * ksteps * 32 + lane;
  const v16h* pb1 = (const v16h*)PB + (long long)(tq * 4 + 1) * ksteps * 32 + lane;
  const v16h* pb2 = (const v16h*)PB + (long long)(tq * 4 + 2) * ksteps * 32 + lane;
  const v16h* pb3 = (const v16h*)PB + (long long)(tq * 4 + 3) * ksteps * 32 + lane;

  v8f acc0 = {}, acc1 = {}, acc2 = {}, acc3 = {};
  for (int k = 0; k < ksteps; ++k) {
    v16h a  = pa[(long long)k * 32];
    v16h b0 = pb0[(long long)k * 32];
    v16h b1 = pb1[(long long)k * 32];
    v16h b2 = pb2[(long long)k * 32];
    v16h b3 = pb3[(long long)k * 32];
    acc0 = __builtin_amdgcn_wmma_f32_16x16x32_f16(false, a, false, b0, (short)0, acc0, false, false);
    acc1 = __builtin_amdgcn_wmma_f32_16x16x32_f16(false, a, false, b1, (short)0, acc1, false, false);
    acc2 = __builtin_amdgcn_wmma_f32_16x16x32_f16(false, a, false, b2, (short)0, acc2, false, false);
    acc3 = __builtin_amdgcn_wmma_f32_16x16x32_f16(false, a, false, b3, (short)0, acc3, false, false);
  }

  // C/D layout: VGPR r -> M = tm*16 + r + (lane>=16 ? 8:0); N = tile_n0 + (lane&15)
  int laneHi = lane >> 4, laneLo = lane & 15;
  int rbase = tm * 16 + laneHi * 8;
  int n0 = tq * 64 + laneLo;
#pragma unroll
  for (int j = 0; j < 4; ++j) {
    v8f acc = (j == 0) ? acc0 : (j == 1) ? acc1 : (j == 2) ? acc2 : acc3;
    int col = n0 + j * 16;
    float bv = bias ? bias[col] : 0.f;
#pragma unroll
    for (int r = 0; r < 8; ++r) {
      float v = acc[r] + bv;
      if (relu) v = fmaxf(v, 0.f);
      C[(long long)(rbase + r) * N + col] = v;
    }
  }
}

// =============================== host side ===============================

namespace {
struct Bump {
  char* base; size_t off; size_t cap;
  void* alloc_bytes(size_t bytes) {
    bytes = (bytes + 255) & ~(size_t)255;
    void* p = (void*)(base + off);
    off += bytes;
    return p;
  }
  float* alloc(size_t nfloats) { return (float*)alloc_bytes(nfloats * sizeof(float)); }
  _Float16* alloc_h(size_t nhalf) { return (_Float16*)alloc_bytes(nhalf * sizeof(_Float16)); }
};

// pack f32 operands to f16 WMMA layouts, then WMMA GEMM (Bump by value: scratch auto-freed;
// safe because all kernels serialize on `stream`).
void launch_gemm(hipStream_t s, Bump ws, const float* A, const float* W,
                 const float* bias, float* C, int M, int N, int K, int relu) {
  _Float16* PA = ws.alloc_h((size_t)M * K);
  _Float16* PB = ws.alloc_h((size_t)K * N);
  int ksteps = K / 32;
  hipLaunchKernelGGL(k_pack_a, dim3(gsz((long long)(M / 16) * ksteps * 32)), dim3(256), 0, s, A, PA, M, K);
  hipLaunchKernelGGL(k_pack_b, dim3(gsz((long long)(N / 16) * ksteps * 32)), dim3(256), 0, s, W, PB, N, K);
  int tiles = (M / 16) * (N / 64);
  hipLaunchKernelGGL(k_gemm_wmma, dim3(cdiv_h(tiles, 4)), dim3(128), 0, s,
                     PA, PB, bias, C, M, N, K, relu);
}

// DGL bipartite GATConv (allow_zero_in_degree, leaky_relu 0.2 attention)
void run_gat(hipStream_t s,
             const float* xs, int ns, int ks,
             const float* xd, int nd, int kd,
             const int* es, const int* ed, int E,
             const float* Wsrc, const float* Wdst,
             const float* al, const float* ar, const float* bias,
             int fo, float* out, Bump ws /* by value: scratch auto-freed */) {
  const int HF = NH * fo;
  float* fs  = ws.alloc((size_t)ns * HF);
  float* fd  = ws.alloc((size_t)nd * HF);
  float* el  = ws.alloc((size_t)ns * NH);
  float* er  = ws.alloc((size_t)nd * NH);
  float* ee  = ws.alloc((size_t)E * NH);
  float* ex  = ws.alloc((size_t)E * NH);
  float* m   = ws.alloc((size_t)nd * NH);
  float* den = ws.alloc((size_t)nd * NH);

  launch_gemm(s, ws, xs, Wsrc, nullptr, fs, ns, HF, ks, 0);
  launch_gemm(s, ws, xd, Wdst, nullptr, fd, nd, HF, kd, 0);
  hipLaunchKernelGGL(k_attn_dot, dim3(gsz((long long)ns * NH)), dim3(256), 0, s, fs, al, el, ns, fo);
  hipLaunchKernelGGL(k_attn_dot, dim3(gsz((long long)nd * NH)), dim3(256), 0, s, fd, ar, er, nd, fo);
  hipLaunchKernelGGL(k_fill, dim3(gsz((long long)nd * NH)), dim3(256), 0, s,
                     m, -__builtin_inff(), (long long)nd * NH);
  hipLaunchKernelGGL(k_edge_logits, dim3(gsz((long long)E * NH)), dim3(256), 0, s,
                     el, er, es, ed, ee, m, E);
  hipLaunchKernelGGL(k_fix_m, dim3(gsz((long long)nd * NH)), dim3(256), 0, s, m, (long long)nd * NH);
  hipLaunchKernelGGL(k_fill, dim3(gsz((long long)nd * NH)), dim3(256), 0, s, den, 0.f, (long long)nd * NH);
  hipLaunchKernelGGL(k_edge_exp, dim3(gsz((long long)E * NH)), dim3(256), 0, s, ee, m, ed, ex, den, E);
  hipLaunchKernelGGL(k_bias_bcast4, dim3(gsz((long long)nd * HF / 4)), dim3(256), 0, s,
                     out, bias, (long long)nd * HF / 4, HF / 4);
  hipLaunchKernelGGL(k_edge_aggr4, dim3(gsz((long long)E * NH * (fo / 4))), dim3(256), 0, s,
                     ex, den, fs, es, ed, out, E, fo);
}
} // namespace

extern "C" void kernel_launch(void* const* d_in, const int* in_sizes, int n_in,
                              void* d_out, int out_size, void* d_ws, size_t ws_size,
                              hipStream_t stream) {
  // ---- inputs in setup_inputs() dict order ----
  const int* user_ids     = (const int*)d_in[0];
  const int* usem_ids     = (const int*)d_in[1];
  const int* item_ids     = (const int*)d_in[2];
  const int* isem_ids     = (const int*)d_in[3];
  const int* us_src       = (const int*)d_in[4];
  const int* us_dst       = (const int*)d_in[5];
  const int* is_src       = (const int*)d_in[6];
  const int* is_dst       = (const int*)d_in[7];
  const int* ui_src       = (const int*)d_in[8];
  const int* ui_dst       = (const int*)d_in[9];
  const int* user_out_idx = (const int*)d_in[10];
  const float* user_table = (const float*)d_in[11];
  const float* usem_table = (const float*)d_in[12];
  const float* item_table = (const float*)d_in[13];
  const float* isem_table = (const float*)d_in[14];
  // relations r1..r5, 5 tensors each: Wsrc, Wdst, al, ar, b
  const float* R[5][5];
  for (int r = 0; r < 5; ++r)
    for (int t = 0; t < 5; ++t)
      R[r][t] = (const float*)d_in[15 + r * 5 + t];
  const float* Wm = (const float*)d_in[40];
  const float* bm = (const float*)d_in[41];
  float* out = (float*)d_out;

  if (ws_size < (size_t)336 * 1024 * 1024) return; // need ~330 MB scratch

  Bump ws{(char*)d_ws, 0, ws_size};
  // persistent intermediates
  float* u_h = ws.alloc((size_t)NUV * NH * FHV);   // [NU,128]
  float* i_h = ws.alloc((size_t)NIV * NH * FHV);   // [NI,128]
  float* h5  = ws.alloc((size_t)NUV * NH * DE);    // [NU,256]
  float* hb  = ws.alloc((size_t)BB * NH * DE);     // [B,256]
  size_t mark = ws.off;

  // ================= user branch =================
  {
    float* user_emb = ws.alloc((size_t)NUV * DE);
    float* u_quant  = ws.alloc((size_t)NUSV * DE);
    float* u_n      = ws.alloc((size_t)NUSV * NH * FHV);
    float* cat_u    = ws.alloc((size_t)NUSV * (NH * FHV + DE));
    hipLaunchKernelGGL(k_gather4, dim3(gsz((long long)NUV * DE / 4)), dim3(256), 0, stream,
                       user_table, user_ids, user_emb, NUV, DE / 4);
    hipLaunchKernelGGL(k_gather4, dim3(gsz((long long)NUSV * DE / 4)), dim3(256), 0, stream,
                       usem_table, usem_ids, u_quant, NUSV, DE / 4);
    // r1: user -shares-> u_sem
    run_gat(stream, user_emb, NUV, DE, u_quant, NUSV, DE, us_src, us_dst, EUS,
            R[0][0], R[0][1], R[0][2], R[0][3], R[0][4], FHV, u_n, ws);
    hipLaunchKernelGGL(k_concat4, dim3(gsz((long long)NUSV * (NH * FHV + DE) / 4)), dim3(256), 0, stream,
                       u_n, u_quant, cat_u, NUSV, NH * FHV / 4, DE / 4);
    // r4: u_sem -denotes-> user (reversed edges)
    run_gat(stream, cat_u, NUSV, NH * FHV + DE, user_emb, NUV, DE, us_dst, us_src, EUS,
            R[3][0], R[3][1], R[3][2], R[3][3], R[3][4], FHV, u_h, ws);
  }
  ws.off = mark;

  // ================= item branch =================
  {
    float* item_emb = ws.alloc((size_t)NIV * DE);
    float* i_quant  = ws.alloc((size_t)NISV * DE);
    float* i_n      = ws.alloc((size_t)NISV * NH * FHV);
    float* cat_i    = ws.alloc((size_t)NISV * (NH * FHV + DE));
    hipLaunchKernelGGL(k_gather4, dim3(gsz((long long)NIV * DE / 4)), dim3(256), 0, stream,
                       item_table, item_ids, item_emb, NIV, DE / 4);
    hipLaunchKernelGGL(k_gather4, dim3(gsz((long long)NISV * DE / 4)), dim3(256), 0, stream,
                       isem_table, isem_ids, i_quant, NISV, DE / 4);
    // r2: item -shares-> i_sem
    run_gat(stream, item_emb, NIV, DE, i_quant, NISV, DE, is_src, is_dst, EIS,
            R[1][0], R[1][1], R[1][2], R[1][3], R[1][4], FHV, i_n, ws);
    hipLaunchKernelGGL(k_concat4, dim3(gsz((long long)NISV * (NH * FHV + DE) / 4)), dim3(256), 0, stream,
                       i_n, i_quant, cat_i, NISV, NH * FHV / 4, DE / 4);
    // r3: i_sem -denotes-> item (reversed edges)
    run_gat(stream, cat_i, NISV, NH * FHV + DE, item_emb, NIV, DE, is_dst, is_src, EIS,
            R[2][0], R[2][1], R[2][2], R[2][3], R[2][4], FHV, i_h, ws);
  }
  ws.off = mark;

  // ================= r5: item -clicked_by-> user =================
  run_gat(stream, i_h, NIV, NH * FHV, u_h, NUV, NH * FHV, ui_src, ui_dst, EUI,
          R[4][0], R[4][1], R[4][2], R[4][3], R[4][4], DE, h5, ws);
  ws.off = mark;

  // ================= batch gather + user_gnn_mapping =================
  hipLaunchKernelGGL(k_gather4, dim3(gsz((long long)BB * NH * DE / 4)), dim3(256), 0, stream,
                     h5, user_out_idx, hb, BB, NH * DE / 4);
  launch_gemm(stream, ws, hb, Wm, bm, out, BB, DE, NH * DE, /*relu=*/1);
}
